// MatchPredictor_1864015806631
// MI455X (gfx1250) — matrix operand order repaired and checked by reading, not compiled, not run on hardware
//
#include <hip/hip_runtime.h>

// ---------------------------------------------------------------------------
// MatchPredictor forward on MI455X (gfx1250, wave32, WMMA f32_16x16x32_f16)
//
// Per block: 64 samples processed as 4 tiles of 16 (weights staged to LDS
// once, amortized over 4 tiles). 5 waves; wave w handles champion slot w for
// both teams. Each (team, slot) runs a 16-row GEMM chain on the matrix pipes
// with K/N zero-padded to WMMA tiles:
//   L1: K 51->64, N 50->64   (8 wmma)
//   L2: K 50->64, N 25->32   (4 wmma)
//   L3: K 25->32, N 10->16   (1 wmma)
// Weights are pre-swizzled in LDS into this code's B-fragment order (k = 16h + s)
// so B fragments are contiguous 32B/lane ds_load_b128 pairs.
// ---------------------------------------------------------------------------

typedef _Float16 v16h __attribute__((ext_vector_type(16)));
typedef _Float16 v8h  __attribute__((ext_vector_type(8)));
typedef float    v8f  __attribute__((ext_vector_type(8)));

#define BATCH    262144
#define SPB_TILE 16          // samples per tile (one WMMA M-tile)
#define TILES    4           // tiles per block
#define SPB      (SPB_TILE * TILES)
#define NWAVES   5
#define NTHREADS (NWAVES * 32)

// Intra-wave LDS producer->consumer ordering: DS ops from one wave complete
// in order; s_wait_dscnt 0 makes all prior LDS writes visible to the wave's
// subsequent LDS reads, and the memory clobber stops compiler reordering.
__device__ __forceinline__ void wave_lds_sync() {
  __builtin_amdgcn_wave_barrier();
  asm volatile("s_wait_dscnt 0x0" ::: "memory");
  __builtin_amdgcn_wave_barrier();
}

// A fragment (16x32 f16, row-major X in LDS, ldh halfs per row).
// Lane L (m = L&15): halfs 0-15 = K k0+0..15 (+16 if L>=16).
__device__ __forceinline__ v16h load_a_frag(const _Float16* X, int ldh,
                                            int k0, int m, int hi) {
  const _Float16* p = X + m * ldh + k0 + (hi ? 16 : 0);
  union { v16h v; v8h h[2]; } u;
  u.h[0] = *(const v8h*)(p);       // ds_load_b128
  u.h[1] = *(const v8h*)(p + 8);  // ds_load_b128
  return u.v;
}

// B fragment: pre-swizzled in LDS as [lane][16 halfs] per 512-half fragment.
__device__ __forceinline__ v16h load_b_frag(const _Float16* frag, int lane) {
  return *(const v16h*)(frag + lane * 16);   // 2x ds_load_b128
}

// Swizzle W (row-major OUT x IN, computing X @ W^T so B = W^T: K=IN, N=OUT)
// into fragment layout: frag = kstep*ntiles + ntile; within a fragment,
// lane = (N&15) + 16*(K&16 ? 1:0), half = K&15  (k order shared with A frags: K = h + 16*(L>=16)).
__device__ __forceinline__ void stage_w(_Float16* dst, const float* W,
                                        int IN, int OUT, int Kpad, int Npad,
                                        int ntiles, int tid) {
  int tot = Kpad * Npad;
  for (int i = tid; i < tot; i += NTHREADS) {
    int K = i / Npad, N = i % Npad;
    float v = (K < IN && N < OUT) ? W[N * IN + K] : 0.0f;
    int lane = (N & 15) + ((K & 16) ? 16 : 0);
    dst[((K >> 5) * ntiles + (N >> 4)) * 512 + lane * 16 + (K & 15)] =
        (_Float16)v;
  }
}

__device__ __forceinline__ void stage_bias(float* dst, const float* b,
                                           int n, int npad, int tid) {
  for (int i = tid; i < npad; i += NTHREADS) dst[i] = (i < n) ? b[i] : 0.0f;
}

__global__ __launch_bounds__(NTHREADS)
void match_mlp_kernel(const float* __restrict__ feat,
                      const float* __restrict__ emb,
                      const float* __restrict__ W1a, const float* __restrict__ b1a,
                      const float* __restrict__ W2a, const float* __restrict__ b2a,
                      const float* __restrict__ W3a, const float* __restrict__ b3a,
                      const float* __restrict__ W1b, const float* __restrict__ b1b,
                      const float* __restrict__ W2b, const float* __restrict__ b2b,
                      const float* __restrict__ W3b, const float* __restrict__ b3b,
                      const float* __restrict__ Wf,  const float* __restrict__ bf,
                      float* __restrict__ out) {
  __shared__ __align__(64) _Float16 sW1[2][8 * 512];  // 2 ksteps x 4 ntiles
  __shared__ __align__(64) _Float16 sW2[2][4 * 512];  // 2 ksteps x 2 ntiles
  __shared__ __align__(64) _Float16 sW3[2][1 * 512];  // 1 x 1
  __shared__ float sB1[2][64], sB2[2][32], sB3[2][16];
  __shared__ float sWf[200], sbf[2];
  __shared__ __align__(64) _Float16 sX[NWAVES][2][16 * 64];  // ping-pong acts
  __shared__ float sX3[SPB_TILE][100];                       // concat buffer

  const int tid = threadIdx.x;

  // ---- stage all weights (tiny) into LDS once; amortized over TILES ----
  stage_w(sW1[0], W1a, 51, 50, 64, 64, 4, tid);
  stage_w(sW1[1], W1b, 51, 50, 64, 64, 4, tid);
  stage_w(sW2[0], W2a, 50, 25, 64, 32, 2, tid);
  stage_w(sW2[1], W2b, 50, 25, 64, 32, 2, tid);
  stage_w(sW3[0], W3a, 25, 10, 32, 16, 1, tid);
  stage_w(sW3[1], W3b, 25, 10, 32, 16, 1, tid);
  stage_bias(sB1[0], b1a, 50, 64, tid);
  stage_bias(sB1[1], b1b, 50, 64, tid);
  stage_bias(sB2[0], b2a, 25, 32, tid);
  stage_bias(sB2[1], b2b, 25, 32, tid);
  stage_bias(sB3[0], b3a, 10, 16, tid);
  stage_bias(sB3[1], b3b, 10, 16, tid);
  for (int i = tid; i < 200; i += NTHREADS) sWf[i] = Wf[i];
  if (tid < 2) sbf[tid] = bf[tid];
  __syncthreads();

  const int wave = tid >> 5;
  const int lane = tid & 31;
  const int nidx = lane & 15;   // N (and M for A-frag rows) index
  const int hi   = lane >> 4;   // upper half-wave

  _Float16* Xa = sX[wave][0];
  _Float16* Xb = sX[wave][1];

  for (int tile = 0; tile < TILES; ++tile) {
    const int sBase = blockIdx.x * SPB + tile * SPB_TILE;

#pragma unroll
    for (int t = 0; t < 2; ++t) {
      const int g = t * 5 + wave;  // global champion slot 0..9

      // ---- build X1 (16 x 64 f16): [emb(50) | glicko(1) | 0-pad] ----
      {
        int r = lane >> 1;
        int s = sBase + r;
        if (t == 0 && tile + 1 < TILES)  // warm next tile's feature rows
          __builtin_prefetch(&feat[(s + SPB_TILE) * 12], 0, 1);
        int ci = (int)feat[s * 12 + 2 + g];
        float gl = feat[s * 12 + t];
        const float* e = emb + ci * 50;
        int c0 = (lane & 1) * 32;
        for (int c = c0; c < c0 + 32; ++c) {
          float v = (c < 50) ? e[c] : ((c == 50) ? gl : 0.0f);
          Xa[(r << 6) + c] = (_Float16)v;
        }
      }
      wave_lds_sync();

      // ---- layer 1: Xa(16x64) x W1(64x64) -> Xb(16x64), bias+ReLU ----
      {
        v16h a0 = load_a_frag(Xa, 64, 0, nidx, hi);
        v16h a1 = load_a_frag(Xa, 64, 32, nidx, hi);
#pragma unroll
        for (int nt = 0; nt < 4; ++nt) {
          v8f acc = {0.f, 0.f, 0.f, 0.f, 0.f, 0.f, 0.f, 0.f};
          acc = __builtin_amdgcn_wmma_f32_16x16x32_f16(
              false, a0, false, load_b_frag(&sW1[t][(0 * 4 + nt) * 512], lane),
              (short)0, acc, false, false);
          acc = __builtin_amdgcn_wmma_f32_16x16x32_f16(
              false, a1, false, load_b_frag(&sW1[t][(1 * 4 + nt) * 512], lane),
              (short)0, acc, false, false);
          float b = sB1[t][nt * 16 + nidx];
#pragma unroll
          for (int j = 0; j < 8; ++j) {
            float v = acc[j] + b;
            Xb[(j + hi * 8) * 64 + nt * 16 + nidx] =
                (_Float16)(v > 0.f ? v : 0.f);
          }
        }
      }
      wave_lds_sync();

      // ---- layer 2: Xb(16x64) x W2(64x32) -> Xa(16x32), bias+ReLU ----
      {
        v16h a0 = load_a_frag(Xb, 64, 0, nidx, hi);
        v16h a1 = load_a_frag(Xb, 64, 32, nidx, hi);
#pragma unroll
        for (int nt = 0; nt < 2; ++nt) {
          v8f acc = {0.f, 0.f, 0.f, 0.f, 0.f, 0.f, 0.f, 0.f};
          acc = __builtin_amdgcn_wmma_f32_16x16x32_f16(
              false, a0, false, load_b_frag(&sW2[t][(0 * 2 + nt) * 512], lane),
              (short)0, acc, false, false);
          acc = __builtin_amdgcn_wmma_f32_16x16x32_f16(
              false, a1, false, load_b_frag(&sW2[t][(1 * 2 + nt) * 512], lane),
              (short)0, acc, false, false);
          float b = sB2[t][nt * 16 + nidx];
#pragma unroll
          for (int j = 0; j < 8; ++j) {
            float v = acc[j] + b;
            Xa[(j + hi * 8) * 32 + nt * 16 + nidx] =
                (_Float16)(v > 0.f ? v : 0.f);
          }
        }
      }
      wave_lds_sync();

      // ---- layer 3: Xa(16x32) x W3(32x16) -> sX3[s][g*10 + 0..9] ----
      {
        v16h a0 = load_a_frag(Xa, 32, 0, nidx, hi);
        v8f acc = {0.f, 0.f, 0.f, 0.f, 0.f, 0.f, 0.f, 0.f};
        acc = __builtin_amdgcn_wmma_f32_16x16x32_f16(
            false, a0, false, load_b_frag(sW3[t], lane),
            (short)0, acc, false, false);
        if (nidx < 10) {
          float b = sB3[t][nidx];
#pragma unroll
          for (int j = 0; j < 8; ++j) {
            float v = acc[j] + b;
            sX3[j + hi * 8][g * 10 + nidx] = (v > 0.f ? v : 0.f);
          }
        }
      }
      wave_lds_sync();  // Xa reused as X1 next team iteration
    }
    __syncthreads();  // all 10 slots of this tile written to sX3

    // ---- final head: out[s,o] = bf[o] + cat(s) . Wf[o,:]  (200 MAC) ----
    if (tid < SPB_TILE * 2) {
      int s = tid >> 1, o = tid & 1;
      const float* wf = &sWf[o * 100];
      const float* x = sX3[s];
      float acc = sbf[o];
#pragma unroll 4
      for (int k = 0; k < 100; ++k) acc = fmaf(x[k], wf[k], acc);
      out[(sBase + s) * 2 + o] = acc;
    }
    __syncthreads();  // sX3 reused next tile
  }
}

extern "C" void kernel_launch(void* const* d_in, const int* in_sizes, int n_in,
                              void* d_out, int out_size, void* d_ws,
                              size_t ws_size, hipStream_t stream) {
  const float* feat = (const float*)d_in[0];
  const float* emb  = (const float*)d_in[1];
  const float* W1a  = (const float*)d_in[2];
  const float* b1a  = (const float*)d_in[3];
  const float* W2a  = (const float*)d_in[4];
  const float* b2a  = (const float*)d_in[5];
  const float* W3a  = (const float*)d_in[6];
  const float* b3a  = (const float*)d_in[7];
  const float* W1b  = (const float*)d_in[8];
  const float* b1b  = (const float*)d_in[9];
  const float* W2b  = (const float*)d_in[10];
  const float* b2b  = (const float*)d_in[11];
  const float* W3b  = (const float*)d_in[12];
  const float* b3b  = (const float*)d_in[13];
  const float* Wf   = (const float*)d_in[14];
  const float* bf   = (const float*)d_in[15];

  dim3 grid(BATCH / SPB);
  match_mlp_kernel<<<grid, NTHREADS, 0, stream>>>(
      feat, emb, W1a, b1a, W2a, b2a, W3a, b3a,
      W1b, b1b, W2b, b2b, W3b, b3b, Wf, bf, (float*)d_out);
}
